// PointPillars_11192684774187
// MI455X (gfx1250) — matrix operand order repaired and checked by reading, not compiled, not run on hardware
//
#include <hip/hip_runtime.h>
#include <cstdint>
#include <cstddef>

// ---------------------------------------------------------------------------
// PointPillars forward for MI455X (gfx1250, wave32, WMMA).
// conv/deconv/head GEMMs -> v_wmma_f32_16x16x32_bf16, f32 accumulate.
// Activations bf16 NHWC. Head weights staged in LDS via async-to-LDS DMA.
// ---------------------------------------------------------------------------

typedef unsigned short bf16_t;                      // raw bf16 storage
typedef __attribute__((ext_vector_type(16))) __bf16 v16bf;
typedef __attribute__((ext_vector_type(8)))  float  v8f;

union Frag {                                        // 32B WMMA A/B fragment
  struct { uint4 lo, hi; } u;
  v16bf v;
};

__device__ __forceinline__ bf16_t f2bf(float f) {
  union { float f; unsigned u; } c; c.f = f;
  unsigned u = c.u;
  unsigned r = (u + 0x7FFFu + ((u >> 16) & 1u)) >> 16;   // RNE
  return (bf16_t)r;
}

// ---------------------------------------------------------------------------
// misc: zero a region with 16B stores
// ---------------------------------------------------------------------------
__global__ void zero16_kernel(uint4* __restrict__ p, long long n) {
  long long i = (long long)blockIdx.x * blockDim.x + threadIdx.x;
  long long s = (long long)gridDim.x * blockDim.x;
  for (; i < n; i += s) p[i] = make_uint4(0u, 0u, 0u, 0u);
}

// ---------------------------------------------------------------------------
// Pillar encoder: one wave per pillar (lane = point index).
// ---------------------------------------------------------------------------
__global__ __launch_bounds__(128)
void encoder_kernel(const float* __restrict__ pillars, const int* __restrict__ coors,
                    const int* __restrict__ npoints, const float* __restrict__ W,
                    const float* __restrict__ es, const float* __restrict__ eb,
                    bf16_t* __restrict__ canvas, int P)
{
  const int wave = threadIdx.x >> 5;
  const int lane = threadIdx.x & 31;
  const int pid  = blockIdx.x * 4 + wave;
  if (pid >= P) return;

  const float* pt = pillars + ((long long)pid * 32 + lane) * 7;
  float x = pt[0], y = pt[1], z = pt[2];
  float f3 = pt[3], f4 = pt[4], f5 = pt[5], f6 = pt[6];

  float sx = x, sy = y, sz = z;
  #pragma unroll
  for (int off = 16; off > 0; off >>= 1) {
    sx += __shfl_xor(sx, off, 32);
    sy += __shfl_xor(sy, off, 32);
    sz += __shfl_xor(sz, off, 32);
  }
  const int   npi = npoints[pid];
  const float inv = 1.0f / (float)npi;
  const float mx = sx * inv, my = sy * inv, mz = sz * inv;

  const int b  = coors[pid * 4 + 0];
  const int cx = coors[pid * 4 + 1];
  const int cy = coors[pid * 4 + 2];
  const float xo = x - ((float)cx * 0.16f + 0.08f);
  const float yo = y - ((float)cy * 0.16f + (0.08f - 39.68f));

  float feat[12] = { xo, yo, z, f3, f4, f5, f6, x - mx, y - my, z - mz, xo, yo };
  if (lane >= npi) {
    #pragma unroll
    for (int c = 0; c < 12; c++) feat[c] = 0.0f;
  }

  const long long base = (((long long)b * 496 + cy) * 432 + cx) * 64;
  for (int o = 0; o < 64; o++) {
    float d = 0.0f;
    #pragma unroll
    for (int c = 0; c < 12; c++) d += feat[c] * W[o * 12 + c];
    float h = d * es[o] + eb[o];
    h = h > 0.0f ? h : 0.0f;
    #pragma unroll
    for (int off = 16; off > 0; off >>= 1) h = fmaxf(h, __shfl_xor(h, off, 32));
    if ((o & 31) == lane) canvas[base + o] = f2bf(h);   // 2 channels per lane
  }
}

// ---------------------------------------------------------------------------
// Weight repack kernels (fp32 -> bf16, WMMA B-fragment lane layout).
// 512 bf16 per 16x16xK32 step = [lane][16]; lane<16: col=lane, K=0..7,16..23;
// lane>=16: col=lane-16, K=8..15,24..31.
// ---------------------------------------------------------------------------
__global__ void pack_conv_kernel(const float* __restrict__ w, bf16_t* __restrict__ dst,
                                 int COUT, int CIN)                 // w: (COUT,CIN,3,3)
{
  const int  KC    = CIN / 32;
  const long long total = (long long)COUT * CIN * 9;
  long long i = (long long)blockIdx.x * blockDim.x + threadIdx.x;
  long long s = (long long)gridDim.x * blockDim.x;
  for (; i < total; i += s) {
    int v = (int)(i & 15);
    int lane = (int)((i >> 4) & 31);
    long long rest = i >> 9;
    int kc = (int)(rest % KC); rest /= KC;
    int t  = (int)(rest % 9);  rest /= 9;
    int ct = (int)rest;
    int n = lane & 15, h = lane >> 4;
    int k = (v < 8) ? h * 8 + v : 16 + h * 8 + (v - 8);
    int cout = ct * 16 + n, cin = kc * 32 + k;
    dst[i] = f2bf(w[((long long)cout * CIN + cin) * 9 + t]);
  }
}

__global__ void pack_neck_kernel(const float* __restrict__ w, bf16_t* __restrict__ dst,
                                 int CIN, int K)                    // w: (CIN,128,K,K)
{
  const int KC = CIN / 32;
  const long long total = (long long)CIN * 128 * K * K;
  long long i = (long long)blockIdx.x * blockDim.x + threadIdx.x;
  long long s = (long long)gridDim.x * blockDim.x;
  for (; i < total; i += s) {
    int v = (int)(i & 15);
    int lane = (int)((i >> 4) & 31);
    long long rest = i >> 9;
    int kc  = (int)(rest % KC); rest /= KC;
    int ct  = (int)(rest % 8);  rest /= 8;
    int tap = (int)rest;
    int n = lane & 15, h = lane >> 4;
    int k = (v < 8) ? h * 8 + v : 16 + h * 8 + (v - 8);
    int cin = kc * 32 + k, cout = ct * 16 + n;
    int ky = tap / K, kx = tap % K;
    dst[i] = f2bf(w[(((long long)cin * 128 + cout) * K + ky) * K + kx]);
  }
}

__global__ void pack_head_kernel(const float* __restrict__ w, bf16_t* __restrict__ dst,
                                 int O, long long total)            // w: (O,384), pad N
{
  long long i = (long long)blockIdx.x * blockDim.x + threadIdx.x;
  long long s = (long long)gridDim.x * blockDim.x;
  for (; i < total; i += s) {
    int v = (int)(i & 15);
    int lane = (int)((i >> 4) & 31);
    long long rest = i >> 9;
    int kc = (int)(rest % 12);
    int nt = (int)(rest / 12);
    int n = lane & 15, h = lane >> 4;
    int k = (v < 8) ? h * 8 + v : 16 + h * 8 + (v - 8);
    int o = nt * 16 + n, cin = kc * 32 + k;
    dst[i] = (o < O) ? f2bf(w[(long long)o * 384 + cin]) : (bf16_t)0;
  }
}

// ---------------------------------------------------------------------------
// Conv 3x3 pad=1 (stride templated), NHWC bf16, implicit GEMM via WMMA.
// Wave tile: 16 pixels x (NACC*16) couts, NACC = COUT/64; 4 waves cover all
// couts, so each A fragment feeds NACC WMMAs from registers.
// grid.x = outH * ceil(outW/16); grid.y = 1; grid.z = batch.
// ---------------------------------------------------------------------------
template <int CIN, int COUT, int STRIDE>
__global__ __launch_bounds__(128)
void conv3x3_kernel(const bf16_t* __restrict__ in, const bf16_t* __restrict__ wpk,
                    const float* __restrict__ scale, const float* __restrict__ bias,
                    bf16_t* __restrict__ out,
                    int inH, int inW, int outH, int outW)
{
  constexpr int KC   = CIN / 32;
  constexpr int NACC = COUT / 64;            // cout tiles per wave
  const int wave = threadIdx.x >> 5;
  const int lane = threadIdx.x & 31;
  const int wtiles = (outW + 15) >> 4;
  const int oy  = blockIdx.x / wtiles;
  const int oxb = (blockIdx.x % wtiles) << 4;
  const int ct0 = wave * NACC;
  const int b   = blockIdx.z;
  const int m    = lane & 15;                // A-layout: lane -> pixel (M)
  const int half = lane >> 4;
  const int ox  = oxb + m;
  const int iy0 = oy * STRIDE - 1;
  const int ix0 = ox * STRIDE - 1;
  const bool mvalid = ox < outW;

  v8f acc[NACC];
  #pragma unroll
  for (int j = 0; j < NACC; j++) acc[j] = (v8f){0.f, 0.f, 0.f, 0.f, 0.f, 0.f, 0.f, 0.f};

  for (int t = 0; t < 9; t++) {
    const int ty = t / 3, tx = t % 3;
    const int iy = iy0 + ty, ix = ix0 + tx;
    const bool v = mvalid && ((unsigned)iy < (unsigned)inH) && ((unsigned)ix < (unsigned)inW);
    const bf16_t* ap = in + (((size_t)b * inH + iy) * inW + ix) * CIN + half * 8;
    if (v) __builtin_prefetch((const void*)(ap + CIN), 0, 1);   // warm next tap pixel
    #pragma unroll
    for (int kc = 0; kc < KC; kc++) {
      Frag a;
      if (v) {
        a.u.lo = *(const uint4*)(ap + kc * 32);        // K = half*8 + 0..7
        a.u.hi = *(const uint4*)(ap + kc * 32 + 16);   // K = 16 + half*8 + 0..7
      } else {
        a.u.lo = make_uint4(0u, 0u, 0u, 0u);
        a.u.hi = make_uint4(0u, 0u, 0u, 0u);
      }
      #pragma unroll
      for (int j = 0; j < NACC; j++) {
        Frag bm;
        const bf16_t* bp = wpk + (((size_t)(ct0 + j) * 9 + t) * KC + kc) * 512 + lane * 16;
        bm.u.lo = *(const uint4*)bp;
        bm.u.hi = *(const uint4*)(bp + 8);
        acc[j] = __builtin_amdgcn_wmma_f32_16x16x32_bf16(false, a.v, false, bm.v,
                                                         (short)0, acc[j], false, false);
      }
    }
  }

  // C/D layout: VGPR r -> M = r + half*8 ; N = lane & 15.
  const int n = lane & 15;
  #pragma unroll
  for (int j = 0; j < NACC; j++) {
    const int ch = (ct0 + j) * 16 + n;
    const float sc = scale[ch], bi = bias[ch];
    #pragma unroll
    for (int r = 0; r < 8; r++) {
      const int oxx = oxb + r + half * 8;
      if (oxx < outW) {
        float vv = acc[j][r] * sc + bi;
        vv = vv > 0.0f ? vv : 0.0f;
        out[(((size_t)b * outH + oy) * outW + oxx) * COUT + ch] = f2bf(vv);
      }
    }
  }
}

// ---------------------------------------------------------------------------
// Neck "deconv" (kernel == stride): per tap (ky,kx) a 1x1 GEMM over input
// pixels; wave = 2 cout tiles (4 waves cover Cout=128 = 8 tiles).
// grid.x = inH * ceil(inW/16); grid.y = K*K (tap); grid.z = batch.
// ---------------------------------------------------------------------------
template <int CIN, int K>
__global__ __launch_bounds__(128)
void neck_kernel(const bf16_t* __restrict__ in, const bf16_t* __restrict__ wpk,
                 const float* __restrict__ scale, const float* __restrict__ bias,
                 bf16_t* __restrict__ feat2d, int inH, int inW, int coff)
{
  constexpr int KC = CIN / 32;
  const int wave = threadIdx.x >> 5;
  const int lane = threadIdx.x & 31;
  const int wtiles = (inW + 15) >> 4;
  const int iy  = blockIdx.x / wtiles;
  const int ixb = (blockIdx.x % wtiles) << 4;
  const int tap = blockIdx.y;
  const int ct0 = wave * 2;
  const int b   = blockIdx.z;
  const int ky = tap / K, kx = tap % K;
  const int m    = lane & 15;
  const int half = lane >> 4;
  const int ix = ixb + m;
  const bool v = ix < inW;

  v8f acc[2];
  acc[0] = (v8f){0.f, 0.f, 0.f, 0.f, 0.f, 0.f, 0.f, 0.f};
  acc[1] = acc[0];
  const bf16_t* ap = in + (((size_t)b * inH + iy) * inW + ix) * CIN + half * 8;
  #pragma unroll
  for (int kc = 0; kc < KC; kc++) {
    Frag a;
    if (v) {
      a.u.lo = *(const uint4*)(ap + kc * 32);
      a.u.hi = *(const uint4*)(ap + kc * 32 + 16);
    } else {
      a.u.lo = make_uint4(0u, 0u, 0u, 0u);
      a.u.hi = make_uint4(0u, 0u, 0u, 0u);
    }
    #pragma unroll
    for (int j = 0; j < 2; j++) {
      Frag bm;
      const bf16_t* bp = wpk + ((size_t)(tap * 8 + ct0 + j) * KC + kc) * 512 + lane * 16;
      bm.u.lo = *(const uint4*)bp;
      bm.u.hi = *(const uint4*)(bp + 8);
      acc[j] = __builtin_amdgcn_wmma_f32_16x16x32_bf16(false, a.v, false, bm.v,
                                                       (short)0, acc[j], false, false);
    }
  }

  const int n = lane & 15;
  #pragma unroll
  for (int j = 0; j < 2; j++) {
    const int ch = (ct0 + j) * 16 + n;
    const float sc = scale[ch], bi = bias[ch];
    #pragma unroll
    for (int r = 0; r < 8; r++) {
      const int ix2 = ixb + r + half * 8;
      if (ix2 < inW) {
        const int oy = iy * K + ky, ox = ix2 * K + kx;
        float vv = acc[j][r] * sc + bi;
        vv = vv > 0.0f ? vv : 0.0f;
        feat2d[(((size_t)b * 248 + oy) * 216 + ox) * 384 + coff + ch] = f2bf(vv);
      }
    }
  }
}

// ---------------------------------------------------------------------------
// Head: 1x1 conv + bias. GEMM M=53568 pixels, K=384, N padded to 16.
// This block's 4 weight tiles (48 KB) are DMA'd into LDS with
// global_load_async_to_lds_b128 (ASYNCcnt), then B fragments come from LDS.
// grid.x = 3348 pixel tiles; grid.y = Ntiles/4; grid.z = batch.
// ---------------------------------------------------------------------------
__global__ __launch_bounds__(128)
void head_kernel(const bf16_t* __restrict__ feat2d, const bf16_t* __restrict__ wpk,
                 const float* __restrict__ bias, float* __restrict__ out, int O)
{
  __shared__ __align__(16) bf16_t smB[4 * 12 * 512];   // 48 KB
  const int wave = threadIdx.x >> 5;
  const int lane = threadIdx.x & 31;
  const int pb = blockIdx.x << 4;
  const int nt = blockIdx.y * 4 + wave;
  const int b  = blockIdx.z;
  const int m    = lane & 15;
  const int half = lane >> 4;
  const int p = pb + m;                    // < 53568 (248*216 = 3348*16)

  // async DMA: 3072 x 16B chunks, 128 threads -> 24 chunks each (no divergence)
  const bf16_t* gsrc = wpk + (size_t)blockIdx.y * (4 * 12 * 512);
  for (int c = threadIdx.x; c < 3072; c += 128) {
    unsigned ldsoff = (unsigned)(uintptr_t)(&smB[c * 8]);
    unsigned long long ga = (unsigned long long)(uintptr_t)(gsrc + c * 8);
    asm volatile("global_load_async_to_lds_b128 %0, %1, off"
                 :: "v"(ldsoff), "v"(ga) : "memory");
  }
  asm volatile("s_wait_asynccnt 0x0" ::: "memory");
  __syncthreads();

  v8f acc = {0.f, 0.f, 0.f, 0.f, 0.f, 0.f, 0.f, 0.f};
  const bf16_t* ap = feat2d + ((size_t)b * 53568 + p) * 384 + half * 8;
  const bf16_t* wb = smB + (size_t)wave * 12 * 512;
  #pragma unroll
  for (int kc = 0; kc < 12; kc++) {
    Frag a;
    a.u.lo = *(const uint4*)(ap + kc * 32);
    a.u.hi = *(const uint4*)(ap + kc * 32 + 16);
    Frag bm;
    const bf16_t* bp = wb + (size_t)kc * 512 + lane * 16;   // ds_load_b128 x2
    bm.u.lo = *(const uint4*)bp;
    bm.u.hi = *(const uint4*)(bp + 8);
    acc = __builtin_amdgcn_wmma_f32_16x16x32_bf16(false, a.v, false, bm.v,
                                                  (short)0, acc, false, false);
  }

  const int n = lane & 15;
  const int o = nt * 16 + n;
  if (o < O) {
    const float bi = bias[o];
    #pragma unroll
    for (int r = 0; r < 8; r++) {
      const int pix = pb + r + half * 8;
      out[((size_t)b * O + o) * 53568 + pix] = acc[r] + bi;
    }
  }
}

// ---------------------------------------------------------------------------
// Host side
// ---------------------------------------------------------------------------
extern "C" void kernel_launch(void* const* d_in, const int* in_sizes, int n_in,
                              void* d_out, int out_size, void* d_ws, size_t ws_size,
                              hipStream_t stream)
{
  (void)n_in; (void)out_size; (void)ws_size;
  auto F = [&](int i) { return (const float*)d_in[i]; };

  // --- bump allocator over d_ws (bf16 activation buffers + packed weights)
  char* wsb = (char*)d_ws;
  size_t off = 0;
  auto alloc = [&](size_t bytes) -> char* {
    off = (off + 255) & ~(size_t)255;
    char* p = wsb + off;
    off += bytes;
    return p;
  };

  bf16_t* canvas = (bf16_t*)alloc(2ull * 496 * 432 * 64 * 2);
  bf16_t* b1X    = (bf16_t*)alloc(2ull * 248 * 216 * 64 * 2);
  bf16_t* b1Y    = (bf16_t*)alloc(2ull * 248 * 216 * 64 * 2);
  bf16_t* c2X    = (bf16_t*)alloc(2ull * 124 * 108 * 128 * 2);
  bf16_t* c2Y    = (bf16_t*)alloc(2ull * 124 * 108 * 128 * 2);
  bf16_t* c3X    = (bf16_t*)alloc(2ull * 62 * 54 * 256 * 2);
  bf16_t* c3Y    = (bf16_t*)alloc(2ull * 62 * 54 * 256 * 2);
  bf16_t* f2d    = (bf16_t*)alloc(2ull * 248 * 216 * 384 * 2);

  // backbone conv layers: (weight input index, COUT, CIN)
  struct CL { int wi, cout, cin; };
  const CL convs[16] = {
    {6, 64, 64},  {9, 64, 64},  {12, 64, 64},  {15, 64, 64},
    {18, 128, 64},{21, 128, 128},{24, 128, 128},{27, 128, 128},{30, 128, 128},{33, 128, 128},
    {36, 256, 128},{39, 256, 256},{42, 256, 256},{45, 256, 256},{48, 256, 256},{51, 256, 256}
  };
  bf16_t* pk[16];
  for (int i = 0; i < 16; i++) {
    long long ne = (long long)convs[i].cout * convs[i].cin * 9;
    pk[i] = (bf16_t*)alloc((size_t)ne * 2);
    pack_conv_kernel<<<dim3((unsigned)((ne + 255) / 256)), 256, 0, stream>>>(
        F(convs[i].wi), pk[i], convs[i].cout, convs[i].cin);
  }
  bf16_t* pkN0 = (bf16_t*)alloc(64ull * 128 * 2);
  bf16_t* pkN1 = (bf16_t*)alloc(128ull * 128 * 4 * 2);
  bf16_t* pkN2 = (bf16_t*)alloc(256ull * 128 * 16 * 2);
  pack_neck_kernel<<<dim3(32), 256, 0, stream>>>(F(54), pkN0, 64, 1);
  pack_neck_kernel<<<dim3(256), 256, 0, stream>>>(F(57), pkN1, 128, 2);
  pack_neck_kernel<<<dim3(2048), 256, 0, stream>>>(F(60), pkN2, 256, 4);

  const long long hTot0 = 36ll * 12 * 512, hTot1 = 28ll * 12 * 512, hTot2 = 8ll * 12 * 512;
  bf16_t* pkH0 = (bf16_t*)alloc((size_t)hTot0 * 2);
  bf16_t* pkH1 = (bf16_t*)alloc((size_t)hTot1 * 2);
  bf16_t* pkH2 = (bf16_t*)alloc((size_t)hTot2 * 2);
  pack_head_kernel<<<dim3((unsigned)((hTot0 + 255) / 256)), 256, 0, stream>>>(F(63), pkH0, 567, hTot0);
  pack_head_kernel<<<dim3((unsigned)((hTot1 + 255) / 256)), 256, 0, stream>>>(F(65), pkH1, 441, hTot1);
  pack_head_kernel<<<dim3((unsigned)((hTot2 + 255) / 256)), 256, 0, stream>>>(F(67), pkH2, 126, hTot2);

  // --- encoder: zero canvas, then scatter pooled pillar features
  zero16_kernel<<<dim3(13392), 256, 0, stream>>>((uint4*)canvas, 3428352ll);
  const int P = in_sizes[2];
  encoder_kernel<<<dim3((unsigned)((P + 3) / 4)), 128, 0, stream>>>(
      F(0), (const int*)d_in[1], (const int*)d_in[2], F(3), F(4), F(5), canvas, P);

  // --- backbone (grid.x = outH * ceil(outW/16), grid.y = 1, grid.z = batch)
  const dim3 g1(248 * 14, 1, 2), g2(124 * 7, 1, 2), g3(62 * 4, 1, 2);
  conv3x3_kernel<64, 64, 2><<<g1, 128, 0, stream>>>(canvas, pk[0],  F(7),  F(8),  b1X, 496, 432, 248, 216);
  conv3x3_kernel<64, 64, 1><<<g1, 128, 0, stream>>>(b1X,   pk[1],  F(10), F(11), b1Y, 248, 216, 248, 216);
  conv3x3_kernel<64, 64, 1><<<g1, 128, 0, stream>>>(b1Y,   pk[2],  F(13), F(14), b1X, 248, 216, 248, 216);
  conv3x3_kernel<64, 64, 1><<<g1, 128, 0, stream>>>(b1X,   pk[3],  F(16), F(17), b1Y, 248, 216, 248, 216);

  conv3x3_kernel<64, 128, 2><<<g2, 128, 0, stream>>>(b1Y, pk[4], F(19), F(20), c2X, 248, 216, 124, 108);
  conv3x3_kernel<128, 128, 1><<<g2, 128, 0, stream>>>(c2X, pk[5], F(22), F(23), c2Y, 124, 108, 124, 108);
  conv3x3_kernel<128, 128, 1><<<g2, 128, 0, stream>>>(c2Y, pk[6], F(25), F(26), c2X, 124, 108, 124, 108);
  conv3x3_kernel<128, 128, 1><<<g2, 128, 0, stream>>>(c2X, pk[7], F(28), F(29), c2Y, 124, 108, 124, 108);
  conv3x3_kernel<128, 128, 1><<<g2, 128, 0, stream>>>(c2Y, pk[8], F(31), F(32), c2X, 124, 108, 124, 108);
  conv3x3_kernel<128, 128, 1><<<g2, 128, 0, stream>>>(c2X, pk[9], F(34), F(35), c2Y, 124, 108, 124, 108);

  conv3x3_kernel<128, 256, 2><<<g3, 128, 0, stream>>>(c2Y, pk[10], F(37), F(38), c3X, 124, 108, 62, 54);
  conv3x3_kernel<256, 256, 1><<<g3, 128, 0, stream>>>(c3X, pk[11], F(40), F(41), c3Y, 62, 54, 62, 54);
  conv3x3_kernel<256, 256, 1><<<g3, 128, 0, stream>>>(c3Y, pk[12], F(43), F(44), c3X, 62, 54, 62, 54);
  conv3x3_kernel<256, 256, 1><<<g3, 128, 0, stream>>>(c3X, pk[13], F(46), F(47), c3Y, 62, 54, 62, 54);
  conv3x3_kernel<256, 256, 1><<<g3, 128, 0, stream>>>(c3Y, pk[14], F(49), F(50), c3X, 62, 54, 62, 54);
  conv3x3_kernel<256, 256, 1><<<g3, 128, 0, stream>>>(c3X, pk[15], F(52), F(53), c3Y, 62, 54, 62, 54);

  // --- neck: block1 out = b1Y, block2 out = c2Y, block3 out = c3Y
  neck_kernel<64, 1><<<dim3(248 * 14, 1, 2), 128, 0, stream>>>(b1Y, pkN0, F(55), F(56), f2d, 248, 216, 0);
  neck_kernel<128, 2><<<dim3(124 * 7, 4, 2), 128, 0, stream>>>(c2Y, pkN1, F(58), F(59), f2d, 124, 108, 128);
  neck_kernel<256, 4><<<dim3(62 * 4, 16, 2), 128, 0, stream>>>(c3Y, pkN2, F(61), F(62), f2d, 62, 54, 256);

  // --- heads (concatenated in d_out, NCHW f32)
  float* out0 = (float*)d_out;
  float* out1 = out0 + 2ll * 567 * 53568;
  float* out2 = out1 + 2ll * 441 * 53568;
  head_kernel<<<dim3(3348, 9, 2), 128, 0, stream>>>(f2d, pkH0, F(64), out0, 567);
  head_kernel<<<dim3(3348, 7, 2), 128, 0, stream>>>(f2d, pkH1, F(66), out1, 441);
  head_kernel<<<dim3(3348, 2, 2), 128, 0, stream>>>(f2d, pkH2, F(68), out2, 126);
}
